// NeuralODE_SAC_Manifold_52467320488099
// MI455X (gfx1250) — matrix operand order repaired and checked by reading, not compiled, not run on hardware
//
#include <hip/hip_runtime.h>

typedef __attribute__((ext_vector_type(16))) _Float16 v16h;
typedef __attribute__((ext_vector_type(8)))  float    v8f;

#define BS    2048
#define NDIM  128
#define HID   256
#define SHD   64
#define TM    16
#define STEPS 8

// packed f16 weight offsets in workspace (element counts)
#define OFF_FW1  0u        // 128x256
#define OFF_FW2  32768u    // 256x256
#define OFF_FW3  98304u    // 256x128
#define OFF_KW   131072u   // 128x128
#define OFF_KWT  147456u   // 128x128 transposed (B[k][n] = kw[n][k])
#define OFF_AMW1 163840u   // 128x64
#define OFF_AMW2 172032u   // 64x64
#define OFF_AMW3 176128u   // 64x128
#define OFF_ALW1 184320u   // 128x64
#define OFF_ALW2 192512u   // 64x64
#define OFF_ALW3 196608u   // 64x128

__device__ inline float softplusf(float x) {
    return fmaxf(x, 0.0f) + log1pf(expf(-fabsf(x)));
}
__device__ inline float sanitizef(float v) {
    return (__builtin_isnan(v) || __builtin_isinf(v)) ? 0.0f : v;
}

// ---- WMMA fragment loaders (wave32, 16x16x32 f16) -------------------------
// A 16x32 (MxK) fragment: lane m=lane&15; K chunk base = kb*32 + (lane>=16 ? 8:0);
// elements 0..7 = K base..base+7, elements 8..15 = K base+16..base+23.
__device__ inline v16h load_A(const _Float16* A, int stride, int lane, int kb) {
    const int m = lane & 15;
    const int kbase = kb * 32 + ((lane & 16) ? 8 : 0);
    const _Float16* p = A + m * stride + kbase;
    v16h a;
#pragma unroll
    for (int e = 0; e < 8; ++e) { a[e] = p[e]; a[e + 8] = p[e + 16]; }
    return a;
}

// B 32x16 (KxN) fragment, pre-swizzled in memory: per (nt,kb,lane) 16 contiguous halves.
__device__ inline v16h load_B(const _Float16* Bp, int nt, int kb, int KB, int lane) {
    const _Float16* p = Bp + ((size_t)((nt * KB + kb) * 32 + lane) << 4);
    v16h b;
#pragma unroll
    for (int e = 0; e < 16; ++e) b[e] = p[e];
    return b;
}

__device__ inline v8f gemm_acc(const _Float16* A, int aStride, int KB,
                               const _Float16* Bp, int nt, int lane) {
    v8f c = {};
#pragma unroll
    for (int kb = 0; kb < KB; ++kb) {
        v16h a = load_A(A, aStride, lane, kb);
        v16h b = load_B(Bp, nt, kb, KB, lane);
        c = __builtin_amdgcn_wmma_f32_16x16x32_f16(false, a, false, b,
                                                   (short)0, c, false, false);
    }
    return c;
}

// C/D 16x16 f32: lane n=lane&15, VGPR v -> row v + (lane>=16 ? 8:0).
// ACT: 0=none 1=tanh 2=softplus 3=relu. dst/bias pre-offset to tile column base.
template <int ACT>
__device__ inline void store_f16t(v8f c, _Float16* dst, int rowStride,
                                  const float* bias, int lane) {
    const int n = lane & 15;
    const int rb = (lane & 16) ? 8 : 0;
    const float b = bias ? bias[n] : 0.0f;
#pragma unroll
    for (int v = 0; v < 8; ++v) {
        float x = c[v] + b;
        if (ACT == 1) x = tanhf(x);
        if (ACT == 2) x = softplusf(x);
        if (ACT == 3) x = fmaxf(x, 0.0f);
        dst[(rb + v) * rowStride + n] = (_Float16)x;
    }
}
template <int ACT>
__device__ inline void store_f32t(v8f c, float* dst, int rowStride,
                                  const float* bias, int lane) {
    const int n = lane & 15;
    const int rb = (lane & 16) ? 8 : 0;
    const float b = bias ? bias[n] : 0.0f;
#pragma unroll
    for (int v = 0; v < 8; ++v) {
        float x = c[v] + b;
        if (ACT == 1) x = tanhf(x);
        if (ACT == 2) x = softplusf(x);
        if (ACT == 3) x = fmaxf(x, 0.0f);
        dst[(rb + v) * rowStride + n] = x;
    }
}

// ---- weight pack: f32 row-major [K][N] -> f16 in B-fragment order ----------
__global__ void pack_b_kernel(const float* __restrict__ src, _Float16* __restrict__ dst,
                              int K, int N, int transpose) {
    int t = blockIdx.x * blockDim.x + threadIdx.x;
    if (t >= K * N) return;
    int e    = t & 15;
    int lane = (t >> 4) & 31;
    int r    = t >> 9;
    int KB   = K >> 5;
    int kb   = r % KB;
    int nt   = r / KB;
    int k = kb * 32 + ((lane & 16) ? 8 : 0) + (e & 7) + ((e & 8) ? 16 : 0);
    int n = nt * 16 + (lane & 15);
    float v = transpose ? src[n * K + k] : src[k * N + n];   // square when transposed
    dst[t] = (_Float16)v;
}

// ---- one Euler step: 16 batch rows per block, 8 wave32 ---------------------
__global__ __launch_bounds__(256) void
ode_step_kernel(const float* __restrict__ yprev, float* __restrict__ ynext,
                const float* __restrict__ noise_s, const float* __restrict__ tvec,
                const _Float16* __restrict__ wsp,
                const float* __restrict__ fb1, const float* __restrict__ fb2,
                const float* __restrict__ fb3,
                const float* __restrict__ amb1, const float* __restrict__ amb2,
                const float* __restrict__ amb3,
                const float* __restrict__ alb1, const float* __restrict__ alb2,
                const float* __restrict__ alb3) {
    __shared__ alignas(16) _Float16 bufA[TM * HID];   // x -> h2 -> uf
    __shared__ alignas(16) _Float16 bufB[TM * HID];   // h1 ; later reused as f32 Jg
    __shared__ alignas(16) _Float16 bufC[TM * NDIM];  // am1|al1 -> ug
    __shared__ alignas(16) _Float16 bufD[TM * NDIM];  // am2|al2
    __shared__ alignas(16) float kf[TM * NDIM];       // k_out
    __shared__ alignas(16) float ff[TM * NDIM];       // f_out
    __shared__ alignas(16) float gg[TM * NDIM];       // m_pre -> g_out
    __shared__ alignas(16) float jfb[TM * NDIM];      // ls_pre -> Jf
    __shared__ float scl[TM];
    float* jg = reinterpret_cast<float*>(bufB);

    const int tid  = threadIdx.x;
    const int lane = tid & 31;
    const int wave = tid >> 5;
    const int row0 = blockIdx.x * TM;
    const float dt = tvec[1] - tvec[0];

    // stage 1: x tile -> f16
    for (int i = tid; i < TM * NDIM; i += 256)
        bufA[i] = (_Float16)yprev[(row0 + (i >> 7)) * NDIM + (i & 127)];
    __syncthreads();

    // stage 2: h1 = tanh(x@fw1+fb1); k = tanh(x@kw); actor stage 1 (relu)
    for (int j = 0; j < 2; ++j) {
        int nt = wave * 2 + j;
        v8f c = gemm_acc(bufA, NDIM, 4, wsp + OFF_FW1, nt, lane);
        store_f16t<1>(c, bufB + nt * 16, HID, fb1 + nt * 16, lane);
    }
    {
        v8f c = gemm_acc(bufA, NDIM, 4, wsp + OFF_KW, wave, lane);
        store_f32t<1>(c, kf + wave * 16, NDIM, nullptr, lane);
    }
    {
        const _Float16* bp = (wave < 4) ? (wsp + OFF_AMW1) : (wsp + OFF_ALW1);
        const float* bb    = (wave < 4) ? amb1 : alb1;
        int nt  = wave & 3;
        int col = ((wave < 4) ? 0 : SHD) + nt * 16;
        v8f c = gemm_acc(bufA, NDIM, 4, bp, nt, lane);
        store_f16t<3>(c, bufC + col, NDIM, bb + nt * 16, lane);
    }
    __syncthreads();

    // stage 3: h2 = softplus(h1@fw2+fb2); actor stage 2 (relu)
    for (int j = 0; j < 2; ++j) {
        int nt = wave * 2 + j;
        v8f c = gemm_acc(bufB, HID, 8, wsp + OFF_FW2, nt, lane);
        store_f16t<2>(c, bufA + nt * 16, HID, fb2 + nt * 16, lane);
    }
    {
        const _Float16* bp = (wave < 4) ? (wsp + OFF_AMW2) : (wsp + OFF_ALW2);
        const float* bb    = (wave < 4) ? amb2 : alb2;
        const _Float16* ab = bufC + ((wave < 4) ? 0 : SHD);
        int nt  = wave & 3;
        int col = ((wave < 4) ? 0 : SHD) + nt * 16;
        v8f c = gemm_acc(ab, NDIM, 2, bp, nt, lane);
        store_f16t<3>(c, bufD + col, NDIM, bb + nt * 16, lane);
    }
    __syncthreads();

    // stage 4: f_out = h2@fw3+fb3; m_pre / ls_pre
    {
        v8f c = gemm_acc(bufA, HID, 8, wsp + OFF_FW3, wave, lane);
        store_f32t<0>(c, ff + wave * 16, NDIM, fb3 + wave * 16, lane);
    }
    {
        const _Float16* bp = (wave < 4) ? (wsp + OFF_AMW3) : (wsp + OFF_ALW3);
        const float* bb    = (wave < 4) ? amb3 : alb3;
        const _Float16* ab = bufD + ((wave < 4) ? 0 : SHD);
        float* dst         = (wave < 4) ? gg : jfb;
        for (int j = 0; j < 2; ++j) {
            int nt = (wave & 3) * 2 + j;
            v8f c = gemm_acc(ab, NDIM, 2, bp, nt, lane);
            store_f32t<0>(c, dst + nt * 16, NDIM, bb + nt * 16, lane);
        }
    }
    __syncthreads();

    // stage 5: elementwise actor sample + Jacobian A-operands uf/ug
    for (int i = tid; i < TM * NDIM; i += 256) {
        int r = i >> 7, cc = i & 127;
        float m  = sanitizef(tanhf(gg[i]));
        float sd = softplusf(sanitizef(jfb[i]));
        float ep = noise_s[(row0 + r) * NDIM + cc];
        float g  = tanhf(m + sd * ep);
        gg[i] = g;
        float k  = kf[i];
        float om = 1.0f - k * k;
        bufA[i] = (_Float16)(om * ff[i]);   // uf
        bufC[i] = (_Float16)(om * g);       // ug
    }
    __syncthreads();

    // stage 6: Jf = uf@kwT ; Jg = ug@kwT
    {
        v8f c0 = gemm_acc(bufA, NDIM, 4, wsp + OFF_KWT, wave, lane);
        store_f32t<0>(c0, jfb + wave * 16, NDIM, nullptr, lane);
        v8f c1 = gemm_acc(bufC, NDIM, 4, wsp + OFF_KWT, wave, lane);
        store_f32t<0>(c1, jg + wave * 16, NDIM, nullptr, lane);
    }
    __syncthreads();

    // stage 7: per-row manifold constraints -> scale
    if (tid < TM) {
        float kn2 = 0.f, j2 = 0.f, cd = 0.f;
        for (int c = 0; c < NDIM; ++c) {
            float k = kf[tid * NDIM + c];
            float a = jfb[tid * NDIM + c];
            kn2 += k * k; j2 += a * a; cd += k * jg[tid * NDIM + c];
        }
        float knorm = sqrtf(kn2);
        float kn8 = kn2 * kn2; kn8 *= kn8;            // knorm^8
        float c1 = sqrtf(j2) - 60.0f * kn8 * knorm;   // - ALPHA*knorm^9
        float c2 = cd - 20.0f * kn8 * kn2;            // - BETA *knorm^10
        scl[tid] = ((c1 > 1e-8f) || (c2 < -1e-8f)) ? 0.5f : 1.0f;
    }
    __syncthreads();

    // stage 8: Euler update
    for (int i = tid; i < TM * NDIM; i += 256) {
        int r = i >> 7, cc = i & 127;
        int gi = (row0 + r) * NDIM + cc;
        ynext[gi] = yprev[gi] + scl[r] * (ff[i] + gg[i]) * dt;
    }
}

extern "C" void kernel_launch(void* const* d_in, const int* in_sizes, int n_in,
                              void* d_out, int out_size, void* d_ws, size_t ws_size,
                              hipStream_t stream) {
    (void)in_sizes; (void)n_in; (void)out_size; (void)ws_size;
    const float* y0    = (const float*)d_in[0];
    const float* tvec  = (const float*)d_in[1];
    const float* noise = (const float*)d_in[2];
    const float* fw1 = (const float*)d_in[3];
    const float* fb1 = (const float*)d_in[4];
    const float* fw2 = (const float*)d_in[5];
    const float* fb2 = (const float*)d_in[6];
    const float* fw3 = (const float*)d_in[7];
    const float* fb3 = (const float*)d_in[8];
    const float* amw1 = (const float*)d_in[9];
    const float* amb1 = (const float*)d_in[10];
    const float* amw2 = (const float*)d_in[11];
    const float* amb2 = (const float*)d_in[12];
    const float* amw3 = (const float*)d_in[13];
    const float* amb3 = (const float*)d_in[14];
    const float* alw1 = (const float*)d_in[15];
    const float* alb1 = (const float*)d_in[16];
    const float* alw2 = (const float*)d_in[17];
    const float* alb2 = (const float*)d_in[18];
    const float* alw3 = (const float*)d_in[19];
    const float* alb3 = (const float*)d_in[20];
    const float* kw   = (const float*)d_in[21];

    _Float16* wsp = (_Float16*)d_ws;
    auto pack = [&](const float* src, unsigned off, int K, int N, int tr) {
        int total = K * N;
        pack_b_kernel<<<(total + 255) / 256, 256, 0, stream>>>(src, wsp + off, K, N, tr);
    };
    pack(fw1,  OFF_FW1,  128, 256, 0);
    pack(fw2,  OFF_FW2,  256, 256, 0);
    pack(fw3,  OFF_FW3,  256, 128, 0);
    pack(kw,   OFF_KW,   128, 128, 0);
    pack(kw,   OFF_KWT,  128, 128, 1);
    pack(amw1, OFF_AMW1, 128,  64, 0);
    pack(amw2, OFF_AMW2,  64,  64, 0);
    pack(amw3, OFF_AMW3,  64, 128, 0);
    pack(alw1, OFF_ALW1, 128,  64, 0);
    pack(alw2, OFF_ALW2,  64,  64, 0);
    pack(alw3, OFF_ALW3,  64, 128, 0);

    float* out = (float*)d_out;
    hipMemcpyAsync(out, y0, (size_t)BS * NDIM * sizeof(float),
                   hipMemcpyDeviceToDevice, stream);

    for (int s = 0; s < STEPS; ++s) {
        ode_step_kernel<<<BS / TM, 256, 0, stream>>>(
            out + (size_t)s * BS * NDIM, out + (size_t)(s + 1) * BS * NDIM,
            noise + (size_t)s * BS * NDIM, tvec, wsp,
            fb1, fb2, fb3, amb1, amb2, amb3, alb1, alb2, alb3);
    }
}
